// Coupled_Attention_Module_60825326846201
// MI455X (gfx1250) — compile-verified, matrix operand order checked by reading, not run on hardware
//
#include <hip/hip_runtime.h>
#include <hip/hip_bf16.h>
#include <math.h>
#include <stdint.h>

typedef __attribute__((ext_vector_type(16))) _Float16 v16h;
typedef __attribute__((ext_vector_type(8)))  float    v8f;
typedef __attribute__((ext_vector_type(4)))  unsigned int v4u;
typedef __attribute__((ext_vector_type(8)))  int      v8i;
typedef __attribute__((ext_vector_type(4)))  int      v4i;

#define B_  2
#define S_  1024
#define H_  768
#define NH_ 12
#define DH_ 64
#define M_  (B_ * S_)   /* 2048 rows when (B,S,H) is viewed as (B*S, H) */

#if __has_builtin(__builtin_amdgcn_tensor_load_to_lds)
#define USE_TDM 1
#else
#define USE_TDM 0
#endif

// -------- CDNA5 16-bit fragment k-index mapping (ISA 7.12.2, 16x32 A) -----
// lane&15 selects the row (A) / column (B); lane>>4 selects the K half.
// element e (0..15) of the v16h fragment maps to this K index:
__device__ __forceinline__ int frag_k(int e, int kh) {
  int p = e >> 1, lo = e & 1;
  int base = (p < 4) ? (p * 2) : (16 + (p - 4) * 2);
  return base + kh * 8 + lo;
}

#if USE_TDM
// ---- Tensor Data Mover: 2D f32 tile (tile_w x tile_h) from a strided ------
// ---- row-major tensor into LDS, packed row-major (ISA 08_async_tensor §8) -
__device__ __forceinline__ void tdm_load_2d_f32(const float* gsrc, unsigned lds_off,
                                                unsigned tile_w, unsigned tile_h,
                                                unsigned row_stride /* elements */)
{
  unsigned long long ga = (unsigned long long)(uintptr_t)gsrc;
  v4u g0;
  g0[0] = 1u;                                          // count=1, user descriptor
  g0[1] = lds_off;                                     // lds_addr (bytes)
  g0[2] = (unsigned)(ga & 0xffffffffu);                // global_addr[31:0]
  g0[3] = (unsigned)((ga >> 32) & 0x01ffffffu)         // global_addr[56:32]
        | (2u << 30);                                  // type = 2 ("image")
  v8i g1;
  g1[0] = (int)(2u << 16);                             // data_size=2 -> 4 bytes
  g1[1] = (int)((row_stride & 0xffffu) << 16);         // tensor_dim0[15:0]
  g1[2] = (int)((row_stride >> 16) |
                ((tile_h & 0xffffu) << 16));           // tensor_dim0 hi / tensor_dim1 lo
  g1[3] = (int)((tile_w & 0xffffu) << 16);             // tensor_dim1 hi=0 / tile_dim0
  g1[4] = (int)(tile_h & 0xffffu);                     // tile_dim1 (tile_dim2 = 0)
  g1[5] = (int)row_stride;                             // tensor_dim0_stride[31:0]
  g1[6] = 0;                                           // stride hi / dim1_stride lo
  g1[7] = 0;
  v4i z = {0, 0, 0, 0};
#if defined(__clang_major__) && (__clang_major__ >= 23)
  v8i z8 = {0, 0, 0, 0, 0, 0, 0, 0};
  __builtin_amdgcn_tensor_load_to_lds(g0, g1, z, z, z8, 0);
#else
  __builtin_amdgcn_tensor_load_to_lds(g0, g1, z, z, 0);
#endif
}
__device__ __forceinline__ unsigned lds_off_of(const void* p) {
  return (unsigned)(uintptr_t)p;   // LDS aperture: offset lives in addr[31:0]
}
#endif

// =================== generic GEMM: C = act(A @ W + bias) ===================
// A: (M_, K) row stride lda (lets us read 2H-wide concat buffers)
// W: (K, 768) row-major.  C: row stride ldc (lets us write into cat halves)
// act: 0 none, 1 tanh, 2 sigmoid
__global__ __launch_bounds__(128)
void gemm_wmma(const float* __restrict__ A, int lda,
               const float* __restrict__ W,
               const float* __restrict__ bias,
               float* __restrict__ C, int ldc,
               int K, int act)
{
  __shared__ float Al[16 * 32];
  __shared__ float Bl[32 * 64];
  const int N = H_;
  int tid  = threadIdx.x;
  int wave = tid >> 5, lane = tid & 31;
  int m0 = blockIdx.x * 16;
  int n0 = blockIdx.y * 64;
  int mloc = lane & 15, kh = lane >> 4;
  v8f acc = {};
  for (int k0 = 0; k0 < K; k0 += 32) {
#if USE_TDM
    if (wave == 0) {     // one DMA pair per block per K-step (TENSORcnt-tracked)
      tdm_load_2d_f32(A + (size_t)m0 * lda + k0, lds_off_of(Al), 32, 16, (unsigned)lda);
      tdm_load_2d_f32(W + (size_t)k0 * N + n0, lds_off_of(Bl), 64, 32, (unsigned)N);
      __builtin_amdgcn_s_wait_tensorcnt(0);
    }
#else
    for (int i = tid; i < 16 * 32; i += 128) {
      int r = i >> 5, c = i & 31;
      Al[i] = A[(size_t)(m0 + r) * lda + k0 + c];
    }
    for (int i = tid; i < 32 * 64; i += 128) {
      int r = i >> 6, c = i & 63;
      Bl[i] = W[(size_t)(k0 + r) * N + n0 + c];
    }
#endif
    __syncthreads();
    v16h af, bf;
#pragma unroll
    for (int e = 0; e < 16; ++e) {
      int kk = frag_k(e, kh);
      af[e] = (_Float16)Al[mloc * 32 + kk];
      bf[e] = (_Float16)Bl[kk * 64 + wave * 16 + mloc];
    }
    acc = __builtin_amdgcn_wmma_f32_16x16x32_f16(false, af, false, bf,
                                                 (short)0, acc, false, false);
    __syncthreads();
  }
  int ncol = n0 + wave * 16 + mloc;
  float bv = bias ? bias[ncol] : 0.0f;
#pragma unroll
  for (int r = 0; r < 8; ++r) {
    int m = m0 + r + 8 * kh;              // C layout: lanes 0-15 -> M=r, 16-31 -> M=8+r
    float o = acc[r] + bv;
    if (act == 1)      o = tanhf(o);
    else if (act == 2) o = 1.0f / (1.0f + __expf(-o));
    C[(size_t)m * ldc + ncol] = o;
  }
}

// =================== flash attention per (b, head) =========================
// grid = (S/128, B*NH), block = 256 (8 waves).  Wave w owns 16 i-rows.
// K/V 32x64 tiles are DMA'd (TDM) into LDS and shared across the block.
__global__ __launch_bounds__(256)
void attn_wmma(const float* __restrict__ q, const float* __restrict__ k,
               const float* __restrict__ v, float* __restrict__ outp)
{
  __shared__ float Kl[32 * 64];
  __shared__ float Vl[32 * 64];
  __shared__ _Float16 Pl[8][16 * 32];
  const float scale = 0.125f;                       // 1/sqrt(64)
  int bh = blockIdx.y;
  int b = bh / NH_, h = bh % NH_;
  int tid = threadIdx.x, wave = tid >> 5, lane = tid & 31;
  int i0 = blockIdx.x * 128 + wave * 16;
  int mloc = lane & 15, kh = lane >> 4;

  const float* qrow = q + ((size_t)(b * S_ + i0 + mloc)) * H_ + h * DH_;
  v16h qa0, qa1;                                    // q fragment, K=0..31 / 32..63
#pragma unroll
  for (int e = 0; e < 16; ++e) {
    int kk = frag_k(e, kh);
    qa0[e] = (_Float16)(qrow[kk] * scale);
    qa1[e] = (_Float16)(qrow[32 + kk] * scale);
  }

  v8f o0 = {}, o1 = {}, o2 = {}, o3 = {};           // 16x64 f32 accumulator
  float mrow[8], lrow[8];
#pragma unroll
  for (int r = 0; r < 8; ++r) { mrow[r] = -1e30f; lrow[r] = 0.0f; }

  for (int j0 = 0; j0 < S_; j0 += 32) {
#if USE_TDM
    if (wave == 0) {
      tdm_load_2d_f32(k + ((size_t)(b * S_ + j0)) * H_ + h * DH_, lds_off_of(Kl), 64, 32, H_);
      tdm_load_2d_f32(v + ((size_t)(b * S_ + j0)) * H_ + h * DH_, lds_off_of(Vl), 64, 32, H_);
      __builtin_amdgcn_s_wait_tensorcnt(0);
    }
#else
    for (int i = tid; i < 32 * 64; i += 256) {
      int r = i >> 6, c = i & 63;
      size_t g = ((size_t)(b * S_ + j0 + r)) * H_ + h * DH_ + c;
      Kl[i] = k[g];
      Vl[i] = v[g];
    }
#endif
    __syncthreads();

    // scores S = q(16x64) @ K^T(64x32) as two 16x16 WMMA tiles, K split in 2
    v8f s0 = {}, s1 = {};
    {
      v16h kb;
#pragma unroll
      for (int e = 0; e < 16; ++e) kb[e] = (_Float16)Kl[mloc * 64 + frag_k(e, kh)];
      s0 = __builtin_amdgcn_wmma_f32_16x16x32_f16(false, qa0, false, kb, (short)0, s0, false, false);
#pragma unroll
      for (int e = 0; e < 16; ++e) kb[e] = (_Float16)Kl[mloc * 64 + 32 + frag_k(e, kh)];
      s0 = __builtin_amdgcn_wmma_f32_16x16x32_f16(false, qa1, false, kb, (short)0, s0, false, false);
#pragma unroll
      for (int e = 0; e < 16; ++e) kb[e] = (_Float16)Kl[(16 + mloc) * 64 + frag_k(e, kh)];
      s1 = __builtin_amdgcn_wmma_f32_16x16x32_f16(false, qa0, false, kb, (short)0, s1, false, false);
#pragma unroll
      for (int e = 0; e < 16; ++e) kb[e] = (_Float16)Kl[(16 + mloc) * 64 + 32 + frag_k(e, kh)];
      s1 = __builtin_amdgcn_wmma_f32_16x16x32_f16(false, qa1, false, kb, (short)0, s1, false, false);
    }

    // online softmax: C-layout row of acc VGPR r == row r+8*kh, spread over 16 lanes
    float p0[8], p1[8];
#pragma unroll
    for (int r = 0; r < 8; ++r) {
      float rm = fmaxf(s0[r], s1[r]);
      for (int off = 1; off < 16; off <<= 1)
        rm = fmaxf(rm, __shfl_xor(rm, off, 32));
      float mnew = fmaxf(mrow[r], rm);
      float corr = __expf(mrow[r] - mnew);
      p0[r] = __expf(s0[r] - mnew);
      p1[r] = __expf(s1[r] - mnew);
      float rs = p0[r] + p1[r];
      for (int off = 1; off < 16; off <<= 1)
        rs += __shfl_xor(rs, off, 32);
      mrow[r] = mnew;
      lrow[r] = lrow[r] * corr + rs;
      o0[r] *= corr; o1[r] *= corr; o2[r] *= corr; o3[r] *= corr;
    }

    // re-layout P (C layout) -> A fragment via per-wave LDS staging
    _Float16* pl = &Pl[wave][0];
#pragma unroll
    for (int r = 0; r < 8; ++r) {
      int m = r + 8 * kh;
      pl[m * 32 + mloc]      = (_Float16)p0[r];
      pl[m * 32 + 16 + mloc] = (_Float16)p1[r];
    }
    __asm__ volatile("s_wait_dscnt 0" ::: "memory");  // wave-local LDS ordering
    __builtin_amdgcn_wave_barrier();
    v16h pf;
#pragma unroll
    for (int e = 0; e < 16; ++e) pf[e] = pl[mloc * 32 + frag_k(e, kh)];

    // O += P(16x32) @ V(32x64), four 16-wide N chunks
    v16h vb;
#pragma unroll
    for (int e = 0; e < 16; ++e) vb[e] = (_Float16)Vl[frag_k(e, kh) * 64 + mloc];
    o0 = __builtin_amdgcn_wmma_f32_16x16x32_f16(false, pf, false, vb, (short)0, o0, false, false);
#pragma unroll
    for (int e = 0; e < 16; ++e) vb[e] = (_Float16)Vl[frag_k(e, kh) * 64 + 16 + mloc];
    o1 = __builtin_amdgcn_wmma_f32_16x16x32_f16(false, pf, false, vb, (short)0, o1, false, false);
#pragma unroll
    for (int e = 0; e < 16; ++e) vb[e] = (_Float16)Vl[frag_k(e, kh) * 64 + 32 + mloc];
    o2 = __builtin_amdgcn_wmma_f32_16x16x32_f16(false, pf, false, vb, (short)0, o2, false, false);
#pragma unroll
    for (int e = 0; e < 16; ++e) vb[e] = (_Float16)Vl[frag_k(e, kh) * 64 + 48 + mloc];
    o3 = __builtin_amdgcn_wmma_f32_16x16x32_f16(false, pf, false, vb, (short)0, o3, false, false);
    __syncthreads();
  }

#pragma unroll
  for (int r = 0; r < 8; ++r) {
    float inv = 1.0f / lrow[r];
    size_t base = ((size_t)(b * S_ + i0 + r + 8 * kh)) * H_ + h * DH_;
    outp[base + mloc]      = o0[r] * inv;
    outp[base + 16 + mloc] = o1[r] * inv;
    outp[base + 32 + mloc] = o2[r] * inv;
    outp[base + 48 + mloc] = o3[r] * inv;
  }
}

// ========================= small helper kernels ============================
__global__ void reduce_mean_k(const float* __restrict__ dv, float* __restrict__ dm) {
  int idx = blockIdx.x * blockDim.x + threadIdx.x;
  if (idx >= B_ * H_) return;
  int b = idx / H_, c = idx % H_;
  float s = 0.0f;
  for (int t = 0; t < S_; ++t) s += dv[((size_t)(b * S_ + t)) * H_ + c];
  dm[idx] = s * (1.0f / (float)S_);
}

// diff_mean(B,H) @ WD(H,H), tanh — tiny (M=2), scalar kernel
__global__ void small_gemm_tanh_k(const float* __restrict__ dm,
                                  const float* __restrict__ W,
                                  float* __restrict__ outv) {
  int idx = blockIdx.x * blockDim.x + threadIdx.x;
  if (idx >= B_ * H_) return;
  int b = idx / H_, n = idx % H_;
  float s = 0.0f;
  for (int kk = 0; kk < H_; ++kk) s += dm[b * H_ + kk] * W[(size_t)kk * H_ + n];
  outv[idx] = tanhf(s);
}

// broadcast a (B,H) vector into one half of the (B*S, 2H) cat buffer
__global__ void bcast_cat_k(const float* __restrict__ src, float* __restrict__ cat, int coloff) {
  int idx = blockIdx.x * blockDim.x + threadIdx.x;
  if (idx >= M_ * H_) return;
  int row = idx / H_, c = idx % H_;
  int b = row / S_;
  cat[(size_t)row * (2 * H_) + coloff + c] = src[b * H_ + c];
}

__global__ void copy_cat_k(const float* __restrict__ src, float* __restrict__ cat, int coloff) {
  int idx = blockIdx.x * blockDim.x + threadIdx.x;
  if (idx >= M_ * H_) return;
  int row = idx / H_, c = idx % H_;
  cat[(size_t)row * (2 * H_) + coloff + c] = src[(size_t)row * H_ + c];
}

// softmax over the sequence axis (axis=1) of t(B,S,H), times a multiplier:
// bmode==0: mul is (B,H) broadcast over S; bmode==1: mul is (B,S,H)
__global__ void colsoftmax_k(const float* __restrict__ t, const float* __restrict__ mul,
                             float* __restrict__ outp, int bmode) {
  int idx = blockIdx.x * blockDim.x + threadIdx.x;
  if (idx >= B_ * H_) return;
  int b = idx / H_, c = idx % H_;
  const float* col = t + (size_t)b * S_ * H_ + c;
  float mx = -1e30f;
  for (int s = 0; s < S_; ++s) mx = fmaxf(mx, col[(size_t)s * H_]);
  float sum = 0.0f;
  for (int s = 0; s < S_; ++s) sum += __expf(col[(size_t)s * H_] - mx);
  float inv = 1.0f / sum;
  for (int s = 0; s < S_; ++s) {
    float p = __expf(col[(size_t)s * H_] - mx) * inv;
    float mv = bmode ? mul[((size_t)(b * S_ + s)) * H_ + c] : mul[b * H_ + c];
    outp[((size_t)(b * S_ + s)) * H_ + c] = p * mv;
  }
}

// out[row] = sigmoid( a[row,:] . w[0:H] + b2[row,:] . w[H:2H] ); one wave/row
__global__ __launch_bounds__(256)
void rowdot_sig_k(const float* __restrict__ a, const float* __restrict__ b2,
                  const float* __restrict__ w, float* __restrict__ outv) {
  int row  = blockIdx.x * 8 + (threadIdx.x >> 5);
  int lane = threadIdx.x & 31;
  float s = 0.0f;
  for (int c = lane; c < H_; c += 32)
    s += a[(size_t)row * H_ + c] * w[c] + b2[(size_t)row * H_ + c] * w[H_ + c];
  for (int off = 16; off; off >>= 1) s += __shfl_xor(s, off, 32);
  if (lane == 0) outv[row] = 1.0f / (1.0f + __expf(-s));
}

__global__ void fusion_k(const float* __restrict__ g, const float* __restrict__ vf,
                         const float* __restrict__ df, float* __restrict__ outp) {
  int idx = blockIdx.x * blockDim.x + threadIdx.x;
  if (idx >= M_ * H_) return;
  float gg = g[idx / H_];
  outp[idx] = gg * vf[idx] + (1.0f - gg) * df[idx];
}

__global__ void final_k(const float* __restrict__ nf, const float* __restrict__ pre,
                        float* __restrict__ outp) {
  int idx = blockIdx.x * blockDim.x + threadIdx.x;
  if (idx >= M_ * H_) return;
  outp[idx] = nf[idx / H_] * tanhf(pre[idx]);
}

// ============================= orchestration ===============================
extern "C" void kernel_launch(void* const* d_in, const int* in_sizes, int n_in,
                              void* d_out, int out_size, void* d_ws, size_t ws_size,
                              hipStream_t stream) {
  (void)in_sizes; (void)n_in; (void)out_size; (void)ws_size;
  const float* x         = (const float*)d_in[0];
  const float* y         = (const float*)d_in[1];
  const float* vq_w      = (const float*)d_in[2];
  const float* vq_b      = (const float*)d_in[3];
  const float* vk_w      = (const float*)d_in[4];
  const float* vk_b      = (const float*)d_in[5];
  const float* vv_w      = (const float*)d_in[6];
  const float* vv_b      = (const float*)d_in[7];
  // d_in[8..11] = dq_w, dq_b, dk_w, dk_b : provably dead (softmax of a
  // j-constant score row is uniform) — intentionally skipped.
  const float* dv_w      = (const float*)d_in[12];
  const float* dv_b      = (const float*)d_in[13];
  const float* van_fc_w  = (const float*)d_in[14];
  const float* van_fc_b  = (const float*)d_in[15];
  const float* WD_w      = (const float*)d_in[16];
  const float* d_theta_w = (const float*)d_in[17];
  const float* d_theta_b = (const float*)d_in[18];
  const float* WV_w      = (const float*)d_in[19];
  const float* diff_fc_w = (const float*)d_in[20];
  const float* diff_fc_b = (const float*)d_in[21];
  const float* v_gamma_w = (const float*)d_in[22];
  const float* v_gamma_b = (const float*)d_in[23];
  const float* diff_out_w= (const float*)d_in[24];
  const float* diff_out_b= (const float*)d_in[25];
  const float* van_out_w = (const float*)d_in[26];
  const float* van_out_b = (const float*)d_in[27];
  const float* diff_fus_w= (const float*)d_in[28];
  const float* diff_fus_b= (const float*)d_in[29];
  const float* van_fus_w = (const float*)d_in[30];
  const float* van_fus_b = (const float*)d_in[31];
  const float* gate_w    = (const float*)d_in[32];
  const float* nf_w      = (const float*)d_in[33];
  const float* nf_b      = (const float*)d_in[34];
  const float* nf_out_w  = (const float*)d_in[35];
  const float* final_w   = (const float*)d_in[36];
  const float* final_b   = (const float*)d_in[37];

  float* W = (float*)d_ws;
  const size_t F = (size_t)M_ * H_;
  float* q      = W + 0 * F;
  float* kbuf   = W + 1 * F;
  float* vbuf   = W + 2 * F;
  float* van    = W + 3 * F;    // van_vector (live long)
  float* tmp    = W + 4 * F;    // dv / t1 / t2 / nf_in
  float* dth    = W + 5 * F;    // d_theta
  float* agam   = W + 6 * F;    // a_gamma
  float* dob    = W + 7 * F;    // diff_out, later final pre-activation
  float* vob    = W + 8 * F;    // van_out
  float* dfus   = W + 9 * F;    // diff_fusion
  float* vfus   = W + 10 * F;   // van_fusion
  float* fus    = W + 11 * F;   // fusion
  float* cat    = W + 12 * F;   // (B*S, 2H) concat staging (2F)
  float* dm     = W + 14 * F;            // diff_mean (B,H)
  float* dwd    = dm + B_ * H_;          // tanh(diff_mean @ WD) (B,H)
  float* gate   = dwd + B_ * H_;         // (B*S)
  float* nfv    = gate + M_;             // (B*S)

  dim3 gblk(128);
  dim3 ggrid(M_ / 16, H_ / 64);
  dim3 eblk(256);
  dim3 egrid((M_ * H_ + 255) / 256);

  auto G = [&](const float* A, int lda, const float* Wm, const float* bias,
               float* C, int ldc, int K, int act) {
    hipLaunchKernelGGL(gemm_wmma, ggrid, gblk, 0, stream, A, lda, Wm, bias, C, ldc, K, act);
  };

  // ---- projections (WMMA GEMMs, TDM-staged tiles) ----
  G(x, H_, vq_w, vq_b, q,    H_, H_, 0);
  G(y, H_, vk_w, vk_b, kbuf, H_, H_, 0);
  G(y, H_, vv_w, vv_b, vbuf, H_, H_, 0);
  G(y, H_, dv_w, dv_b, tmp,  H_, H_, 0);   // dv

  // ---- degenerate diff branch: mean over sequence, then @WD ----
  hipLaunchKernelGGL(reduce_mean_k, dim3(6), dim3(256), 0, stream, tmp, dm);
  hipLaunchKernelGGL(small_gemm_tanh_k, dim3(6), dim3(256), 0, stream, dm, WD_w, dwd);

  // ---- vanilla flash attention ----
  hipLaunchKernelGGL(attn_wmma, dim3(S_ / 128, B_ * NH_), dim3(256), 0, stream,
                     q, kbuf, vbuf, van);

  // ---- theta = tanh(cat(diff@WD, van@van_fc + b)); t1 = theta @ d_theta_w + b ----
  G(van, H_, van_fc_w, van_fc_b, cat + H_, 2 * H_, H_, 1);
  hipLaunchKernelGGL(bcast_cat_k, egrid, eblk, 0, stream, dwd, cat, 0);
  G(cat, 2 * H_, d_theta_w, d_theta_b, tmp, H_, 2 * H_, 0);
  hipLaunchKernelGGL(colsoftmax_k, dim3(6), dim3(256), 0, stream, tmp, dm, dth, 0);  // d_theta

  // ---- gamma = tanh(cat(van@WV, d_theta@diff_fc + b)); a_gamma ----
  G(van, H_, WV_w, nullptr, cat, 2 * H_, H_, 1);
  G(dth, H_, diff_fc_w, diff_fc_b, cat + H_, 2 * H_, H_, 1);
  G(cat, 2 * H_, v_gamma_w, v_gamma_b, tmp, H_, 2 * H_, 0);
  hipLaunchKernelGGL(colsoftmax_k, dim3(6), dim3(256), 0, stream, tmp, van, agam, 1); // a_gamma

  // ---- diff_out, van_out ----
  hipLaunchKernelGGL(bcast_cat_k, egrid, eblk, 0, stream, dm, cat, 0);
  hipLaunchKernelGGL(copy_cat_k,  egrid, eblk, 0, stream, dth, cat, H_);
  G(cat, 2 * H_, diff_out_w, diff_out_b, dob, H_, 2 * H_, 1);
  hipLaunchKernelGGL(copy_cat_k, egrid, eblk, 0, stream, van,  cat, 0);
  hipLaunchKernelGGL(copy_cat_k, egrid, eblk, 0, stream, agam, cat, H_);
  G(cat, 2 * H_, van_out_w, van_out_b, vob, H_, 2 * H_, 1);

  // ---- fusions + gate ----
  G(dob, H_, diff_fus_w, diff_fus_b, dfus, H_, H_, 1);
  G(vob, H_, van_fus_w, van_fus_b, vfus, H_, H_, 1);
  hipLaunchKernelGGL(rowdot_sig_k, dim3(M_ / 8), dim3(256), 0, stream, dfus, vfus, gate_w, gate);
  hipLaunchKernelGGL(fusion_k, egrid, eblk, 0, stream, gate, vfus, dfus, fus);

  // ---- nf + final output ----
  G(fus, H_, nf_w, nf_b, tmp, H_, H_, 0);                               // nf_in
  hipLaunchKernelGGL(rowdot_sig_k, dim3(M_ / 8), dim3(256), 0, stream, van, tmp, nf_out_w, nfv);
  G(fus, H_, final_w, final_b, dob, H_, H_, 0);                         // final pre-act
  hipLaunchKernelGGL(final_k, egrid, eblk, 0, stream, nfv, dob, (float*)d_out);
}
